// RSABlock_51591147159722
// MI455X (gfx1250) — matrix operand order repaired and checked by reading, not compiled, not run on hardware
//
#include <hip/hip_runtime.h>

// ---------------- problem constants ----------------
#define BN    4
#define CC    64
#define HH    96
#define WW    96
#define HW    (HH*WW)        // 9216
#define NPIX  (BN*HW)        // 36864
#define KC    576            // 64*9
#define DGR   8
#define CPG   8
#define KK9   9
#define MOFF  216            // 3*DG*KK
#define MOFFP 224            // padded to multiple of 16
#define PH    98             // padded H (+1 each side)
#define PW    98
#define PHW   (PH*PW)        // 9604
#define PPLANE (BN*PHW)      // 38416 per channel
#define NEG_SLOPE 0.2f

typedef __attribute__((ext_vector_type(8)))  float          v8f;
typedef __attribute__((ext_vector_type(4)))  float          v4f;
typedef __attribute__((ext_vector_type(16))) __bf16         v16bf;
typedef __attribute__((ext_vector_type(8)))  unsigned short ushort8;

union Frag16 {
    ushort8        h[2];
    unsigned short u[16];
    v16bf          v;
};

__device__ __forceinline__ unsigned short f32_to_bf16(float f) {
    unsigned int u = __builtin_bit_cast(unsigned int, f);
    u += 0x7fffu + ((u >> 16) & 1u);          // round-to-nearest-even
    return (unsigned short)(u >> 16);
}

// ---------------- weight convert (flat, zero-padded rows) ----------------
__global__ void k_cvt_w(const float* __restrict__ src, unsigned short* __restrict__ dst,
                        int nsrc, int ndst) {
    int i = blockIdx.x * 256 + threadIdx.x;
    if (i < ndst) dst[i] = (i < nsrc) ? f32_to_bf16(src[i]) : (unsigned short)0;
}

// ---------------- f32 copy with zero pad (bias padding) ----------------
__global__ void k_cvt_f32pad(const float* __restrict__ src, float* __restrict__ dst,
                             int nsrc, int ndst) {
    int i = blockIdx.x * 256 + threadIdx.x;
    if (i < ndst) dst[i] = (i < nsrc) ? src[i] : 0.0f;
}

// ---------------- NCHW f32 -> zero-padded bf16 [c][b][98][98] ----------------
__global__ void k_pack_pad(const float* __restrict__ src, unsigned short* __restrict__ dst,
                           int Cch) {
    int t = blockIdx.x * 256 + threadIdx.x;
    if (t >= Cch * PPLANE) return;
    int c   = t / PPLANE;
    int rem = t - c * PPLANE;
    int b   = rem / PHW;
    int pw  = rem - b * PHW;
    int hp  = pw / PW;
    int wp  = pw - hp * PW;
    int h = hp - 1, w = wp - 1;
    unsigned short v = 0;
    if (h >= 0 && h < HH && w >= 0 && w < WW)
        v = f32_to_bf16(src[((size_t)b * Cch + c) * HW + h * WW + w]);
    dst[t] = v;
}

// ---------------- zero a bf16 buffer (fea_pad borders must be 0 every call) ---
__global__ void k_zero_u16(unsigned short* __restrict__ dst, int n) {
    int t = blockIdx.x * 256 + threadIdx.x;
    if (t < n) dst[t] = 0;
}

// ---------------- im2col: padded bf16 [c][b][98][98] -> col[pix][576] ---------
// K = cin*9 + (ky*3+kx); with pad=1, src row = h+ky (ky in 0..2)
__global__ void k_im2col(const unsigned short* __restrict__ padact,
                         unsigned short* __restrict__ colB) {
    int t = blockIdx.x * 256 + threadIdx.x;      // t = p*576 + K
    if (t >= NPIX * KC) return;
    int p   = t / KC;
    int K   = t - p * KC;
    int cin = K / 9;
    int kid = K - cin * 9;
    int dy  = kid / 3;
    int dx  = kid - dy * 3;
    int b   = p / HW;
    int hw  = p - b * HW;
    int h   = hw / WW;
    int w   = hw - h * WW;
    colB[t] = padact[(size_t)cin * PPLANE + b * PHW + (h + dy) * PW + (w + dx)];
}

// ---------------- unified bf16 WMMA GEMM: D = A(MxK) * B(KxN) ----------------
// A  : [Mpad][576] bf16 row-major
// Bt : [pix][576]  bf16 (each pixel's K-column contiguous)
// MODE 0: outM[m][pix] = acc + bias                      (offset conv)
// MODE 1: feaP[padded] = bf16(leaky(acc + bias))         (DCN einsum)
// MODE 2: outNCHW      = acc + bias + resid              (conv1 + residual)
// Each wave: 16 M x 64 N (4 sub-tiles share one A fragment). Block = 8 waves.
template <int MODE>
__global__ void k_gemm_wmma(const unsigned short* __restrict__ A,
                            const unsigned short* __restrict__ Bt,
                            const float* __restrict__ bias,   // padded to >= mbase+16
                            float* __restrict__ outM,
                            unsigned short* __restrict__ feaP,
                            float* __restrict__ outNCHW,
                            const float* __restrict__ resid,
                            int M) {
    const int lane  = threadIdx.x & 31;
    const int wave  = threadIdx.x >> 5;
    const int col   = lane & 15;
    const int hi    = lane >> 4;                 // K-half select
    const int mbase = blockIdx.y * 16;
    const int p0    = (blockIdx.x * 8 + wave) * 64;

    const unsigned short* arow = A + (size_t)(mbase + col) * KC + hi * 16;
    const unsigned short* br0  = Bt + (size_t)(p0 +  0 + col) * KC + hi * 16;
    const unsigned short* br1  = Bt + (size_t)(p0 + 16 + col) * KC + hi * 16;
    const unsigned short* br2  = Bt + (size_t)(p0 + 32 + col) * KC + hi * 16;
    const unsigned short* br3  = Bt + (size_t)(p0 + 48 + col) * KC + hi * 16;

    v8f acc0 = {}, acc1 = {}, acc2 = {}, acc3 = {};

#pragma unroll 2
    for (int K0 = 0; K0 < KC; K0 += 32) {
        // gfx1250 prefetch of the next K-block of each B row (speculative, no cnt)
        __builtin_prefetch(br0 + K0 + 64, 0, 1);
        __builtin_prefetch(br1 + K0 + 64, 0, 1);
        __builtin_prefetch(br2 + K0 + 64, 0, 1);
        __builtin_prefetch(br3 + K0 + 64, 0, 1);

        Frag16 af;
        const ushort8* ap = reinterpret_cast<const ushort8*>(arow + K0);
        af.h[0] = ap[0];                 // K = K0 + hi*16 .. +7
        af.h[1] = ap[1];                 // K = K0 + hi*16 + 8 .. +15

        Frag16 b0, b1, b2, b3;
        const ushort8* bp;
        bp = reinterpret_cast<const ushort8*>(br0 + K0); b0.h[0] = bp[0]; b0.h[1] = bp[1];
        bp = reinterpret_cast<const ushort8*>(br1 + K0); b1.h[0] = bp[0]; b1.h[1] = bp[1];
        bp = reinterpret_cast<const ushort8*>(br2 + K0); b2.h[0] = bp[0]; b2.h[1] = bp[1];
        bp = reinterpret_cast<const ushort8*>(br3 + K0); b3.h[0] = bp[0]; b3.h[1] = bp[1];

        acc0 = __builtin_amdgcn_wmma_f32_16x16x32_bf16(false, af.v, false, b0.v,
                                                       (short)0, acc0, false, false);
        acc1 = __builtin_amdgcn_wmma_f32_16x16x32_bf16(false, af.v, false, b1.v,
                                                       (short)0, acc1, false, false);
        acc2 = __builtin_amdgcn_wmma_f32_16x16x32_bf16(false, af.v, false, b2.v,
                                                       (short)0, acc2, false, false);
        acc3 = __builtin_amdgcn_wmma_f32_16x16x32_bf16(false, af.v, false, b3.v,
                                                       (short)0, acc3, false, false);
    }

    // Unconditional vectorized bias load: this half's 8 rows are contiguous.
    float barr[8];
    {
        const v4f* bp4 = reinterpret_cast<const v4f*>(bias + mbase + hi * 8);
        v4f b0 = bp4[0], b1 = bp4[1];
#pragma unroll
        for (int r = 0; r < 4; ++r) { barr[r] = b0[r]; barr[4 + r] = b1[r]; }
    }

    // epilogue per 16x16 sub-tile; C/D layout: vgpr r -> M = mbase + r + hi*8, N = pcol
    auto store_tile = [&](const v8f& acc, int pcol) {
        const int b  = pcol / HW;
        const int hw = pcol - b * HW;
        const int h  = hw / WW;
        const int w  = hw - h * WW;
        if (MODE == 0) {
#pragma unroll
            for (int r = 0; r < 8; ++r) {
                int mm  = mbase + r + hi * 8;
                float v = acc[r] + barr[r];
                if (mm < M) outM[(size_t)mm * NPIX + pcol] = v;   // guard: store only
            }
        } else if (MODE == 1) {
            size_t base = (size_t)(mbase + hi * 8) * PPLANE +
                          (size_t)b * PHW + (h + 1) * PW + (w + 1);
#pragma unroll
            for (int r = 0; r < 8; ++r) {
                float v = acc[r] + barr[r];
                v = (v >= 0.0f) ? v : NEG_SLOPE * v;
                feaP[base + (size_t)r * PPLANE] = f32_to_bf16(v);
            }
        } else {
            size_t base = ((size_t)b * CC + mbase + hi * 8) * HW + hw;
            float rs[8];
#pragma unroll
            for (int r = 0; r < 8; ++r)                 // hoisted: clause-able loads
                rs[r] = resid[base + (size_t)r * HW];
#pragma unroll
            for (int r = 0; r < 8; ++r)
                outNCHW[base + (size_t)r * HW] = acc[r] + barr[r] + rs[r];
        }
    };
    store_tile(acc0, p0 +  0 + col);
    store_tile(acc1, p0 + 16 + col);
    store_tile(acc2, p0 + 32 + col);
    store_tile(acc3, p0 + 48 + col);
}

// ---------------- deformable bilinear sampler ----------------
// om: [216][NPIX] f32 (oy: 0-71, ox: 72-143, mask: 144-215; chan = g*9+k)
// out: sampT[pix][576] bf16, K = (g*8+c)*9 + k
__global__ void k_sample(const float* __restrict__ x, const float* __restrict__ om,
                         unsigned short* __restrict__ sampT) {
    int t = blockIdx.x * 256 + threadIdx.x;
    if (t >= DGR * KK9 * NPIX) return;
    int gk = t / NPIX;
    int p  = t - gk * NPIX;
    int g  = gk / KK9;
    int k  = gk - g * KK9;
    int dy = k / 3 - 1;
    int dx = k - (k / 3) * 3 - 1;
    int b  = p / HW;
    int hw = p - b * HW;
    int h  = hw / WW;
    int w  = hw - h * WW;

    float oy = om[(size_t)gk * NPIX + p];
    float ox = om[(size_t)(72 + gk) * NPIX + p];
    float mk = om[(size_t)(144 + gk) * NPIX + p];
    mk = 1.0f / (1.0f + expf(-mk));

    float py  = (float)(h + dy) + oy;
    float px  = (float)(w + dx) + ox;
    float y0f = floorf(py), x0f = floorf(px);
    float fy  = py - y0f,   fx  = px - x0f;
    int y0 = (int)y0f, x0 = (int)x0f;
    int y1 = y0 + 1,   x1 = x0 + 1;
    bool vy0 = (y0 >= 0) && (y0 < HH), vy1 = (y1 >= 0) && (y1 < HH);
    bool vx0 = (x0 >= 0) && (x0 < WW), vx1 = (x1 >= 0) && (x1 < WW);
    int y0c = min(max(y0, 0), HH - 1), y1c = min(max(y1, 0), HH - 1);
    int x0c = min(max(x0, 0), WW - 1), x1c = min(max(x1, 0), WW - 1);
    float w00 = (1.f - fy) * (1.f - fx) * ((vy0 && vx0) ? 1.f : 0.f);
    float w01 = (1.f - fy) * fx         * ((vy0 && vx1) ? 1.f : 0.f);
    float w10 = fy * (1.f - fx)         * ((vy1 && vx0) ? 1.f : 0.f);
    float w11 = fy * fx                 * ((vy1 && vx1) ? 1.f : 0.f);

    const float* xb = x + ((size_t)b * CC + g * CPG) * HW;
    int i00 = y0c * WW + x0c, i01 = y0c * WW + x1c;
    int i10 = y1c * WW + x0c, i11 = y1c * WW + x1c;
    unsigned short* outp = sampT + (size_t)p * KC + (g * CPG) * KK9 + k;
#pragma unroll
    for (int c = 0; c < CPG; ++c) {
        const float* xc = xb + (size_t)c * HW;
        float v = w00 * xc[i00] + w01 * xc[i01] + w10 * xc[i10] + w11 * xc[i11];
        outp[c * KK9] = f32_to_bf16(v * mk);
    }
}

// ---------------- host launcher ----------------
extern "C" void kernel_launch(void* const* d_in, const int* in_sizes, int n_in,
                              void* d_out, int out_size, void* d_ws, size_t ws_size,
                              hipStream_t stream) {
    const float* x       = (const float*)d_in[0];
    const float* offset  = (const float*)d_in[1];
    const float* w_off   = (const float*)d_in[2];
    const float* b_off   = (const float*)d_in[3];
    const float* w_dcn   = (const float*)d_in[4];
    const float* b_dcn   = (const float*)d_in[5];
    const float* w_conv1 = (const float*)d_in[6];
    const float* b_conv1 = (const float*)d_in[7];
    float* out = (float*)d_out;

    char* ws = (char*)d_ws;
    // 256B-aligned carve-up with lifetime-based aliasing:
    //   big : col_off -> sampT -> col_fea (sequential lifetimes)
    //   omR : om (f32) -> fea_pad (bf16)  (sequential lifetimes)
    unsigned short* w_off_b  = (unsigned short*)(ws + 0);         // 224*576 bf16 (258,048 B)
    unsigned short* w_dcn_b  = (unsigned short*)(ws + 258048);    //  64*576 bf16 ( 73,728 B)
    unsigned short* w_c1_b   = (unsigned short*)(ws + 331776);    //  64*576 bf16 ( 73,728 B)
    unsigned short* offp     = (unsigned short*)(ws + 405504);    // 64*PPLANE bf16 (4,917,248 B)
    unsigned short* big      = (unsigned short*)(ws + 5322752);   // NPIX*576 bf16 (42,467,328 B)
    char*           omR      = ws + 47790080;                     // 31,850,496 B region
    float*          om       = (float*)omR;                       // 216*NPIX f32
    unsigned short* feapad   = (unsigned short*)omR;              // 64*PPLANE bf16 (aliased)
    float*          b_offpad = (float*)(ws + 79640576);           // 224 f32 (1,024 B slot)
    // total = 79,641,600 bytes

    // ---- stage 0: weight/bias converts + padded offset activations ----
    k_cvt_w<<<(MOFFP * KC + 255) / 256, 256, 0, stream>>>(w_off, w_off_b, MOFF * KC, MOFFP * KC);
    k_cvt_w<<<(CC * KC + 255) / 256, 256, 0, stream>>>(w_dcn, w_dcn_b, CC * KC, CC * KC);
    k_cvt_w<<<(CC * KC + 255) / 256, 256, 0, stream>>>(w_conv1, w_c1_b, CC * KC, CC * KC);
    k_cvt_f32pad<<<1, 256, 0, stream>>>(b_off, b_offpad, MOFF, MOFFP);
    k_pack_pad<<<(CC * PPLANE + 255) / 256, 256, 0, stream>>>(offset, offp, CC);

    // ---- stage 1: offset conv = im2col + GEMM (M=216 pad 224) -> om ----
    k_im2col<<<(NPIX * KC + 255) / 256, 256, 0, stream>>>(offp, big);
    dim3 g1(NPIX / 512, MOFFP / 16);
    k_gemm_wmma<0><<<g1, 256, 0, stream>>>(w_off_b, big, b_offpad, om,
                                           nullptr, nullptr, nullptr, MOFF);

    // ---- stage 2: deformable bilinear sampling -> sampT (in `big`) ----
    k_sample<<<(DGR * KK9 * NPIX + 255) / 256, 256, 0, stream>>>(x, om, big);

    // ---- stage 3: DCN GEMM + bias + LeakyReLU -> fea_pad (aliases dead om) ----
    k_zero_u16<<<(CC * PPLANE + 255) / 256, 256, 0, stream>>>(feapad, CC * PPLANE);
    dim3 g2(NPIX / 512, CC / 16);
    k_gemm_wmma<1><<<g2, 256, 0, stream>>>(w_dcn_b, big, b_dcn, nullptr,
                                           feapad, nullptr, nullptr, CC);

    // ---- stage 4: conv1 = im2col(fea_pad) + GEMM + bias + residual -> d_out ----
    k_im2col<<<(NPIX * KC + 255) / 256, 256, 0, stream>>>(feapad, big);
    dim3 g3(NPIX / 512, CC / 16);
    k_gemm_wmma<2><<<g3, 256, 0, stream>>>(w_c1_b, big, b_conv1, nullptr,
                                           nullptr, out, x, CC);

    (void)in_sizes; (void)n_in; (void)out_size; (void)ws_size;
}
